// Graph_Model_35716948033625
// MI455X (gfx1250) — matrix operand order repaired and checked by reading, not compiled, not run on hardware
//
#include <hip/hip_runtime.h>
#include <hip/hip_bf16.h>

#define F 128
#define HEADS 4
#define NEG_SLOPE 0.2f
#define EPS_A 1e-16f

typedef __attribute__((ext_vector_type(16))) __bf16 v16bf;
typedef __attribute__((ext_vector_type(8)))  __bf16 v8bf;
typedef __attribute__((ext_vector_type(8)))  float  v8f;

__device__ __forceinline__ __bf16 f2bf_rne(float f) {
  unsigned u = __float_as_uint(f);
  unsigned r = (u + 0x7fffu + ((u >> 16) & 1u)) >> 16;
  unsigned short s = (unsigned short)r;
  return __builtin_bit_cast(__bf16, s);
}

__device__ __forceinline__ float lrelu(float v) { return v > 0.f ? v : NEG_SLOPE * v; }

// fp32 -> bf16 elementwise (x matrix)
__global__ void convert_x_kernel(const float* __restrict__ in, __bf16* __restrict__ out, int n) {
  int t = blockIdx.x * blockDim.x + threadIdx.x;
  if (t < n) out[t] = f2bf_rne(in[t]);
}

// W [K=128][N=128] fp32 row-major -> WbT [N][K] bf16 (WMMA-B fragment friendly)
__global__ void convert_w_kernel(const float* __restrict__ W, __bf16* __restrict__ WbT) {
  int t = blockIdx.x * blockDim.x + threadIdx.x; // 16384 threads
  int k = t >> 7, n = t & (F - 1);
  WbT[n * F + k] = f2bf_rne(W[k * F + n]);
}

// zero the accumulation buffer, set segment-max to -inf, segment-sum to 0
__global__ void init_kernel(float* __restrict__ buf, float* __restrict__ mb,
                            float* __restrict__ sb, int N) {
  int t = blockIdx.x * blockDim.x + threadIdx.x;
  if (t < N * F) buf[t] = 0.f;
  if (t < N * HEADS) { mb[t] = -INFINITY; sb[t] = 0.f; }
}

// h = A(bf16, [N,128]) @ BT(bf16, [128 out][128 k]) -> fp32 [N,128]
// one wave computes a 16x64 output slab: 4 accumulators, K=128 in 4 steps of 32
__global__ void __launch_bounds__(256) gemm_wmma_kernel(const __bf16* __restrict__ A,
                                                        const __bf16* __restrict__ BT,
                                                        float* __restrict__ H, int N) {
  int wave = blockIdx.x * (blockDim.x >> 5) + (threadIdx.x >> 5);
  int row_tile = wave >> 1;       // 16 rows each
  int ngrp = wave & 1;            // which 64-column half
  if (row_tile >= (N >> 4)) return;
  int lane = threadIdx.x & 31;
  int m  = lane & 15;             // row within tile (A) / column within tile (B,D)
  int hi = lane >> 4;             // lane-half selector
  v8f acc0 = {}, acc1 = {}, acc2 = {}, acc3 = {};
#if defined(__gfx1250__)
  const __bf16* arow = A + (size_t)(row_tile * 16 + m) * F + (hi << 4);
  #pragma unroll
  for (int ks = 0; ks < 4; ++ks) {
    int k0 = ks * 32;
    // A 16x32 bf16 fragment: lanes 0-15 rows, element i -> k = hi*16 + i
    v8bf alo = *(const v8bf*)(arow + k0);
    v8bf ahi = *(const v8bf*)(arow + k0 + 8);
    v16bf a = __builtin_shufflevector(alo, ahi, 0,1,2,3,4,5,6,7,8,9,10,11,12,13,14,15);
    // B 32x16 fragment from transposed [n][k] layout: lane -> n, element i -> k = k0 + hi*16 + i
    const __bf16* bbase = BT + (size_t)(ngrp * 64 + m) * F + k0 + (hi << 4);
    v16bf b0 = *(const v16bf*)(bbase);
    v16bf b1 = *(const v16bf*)(bbase + 16 * F);
    v16bf b2 = *(const v16bf*)(bbase + 32 * F);
    v16bf b3 = *(const v16bf*)(bbase + 48 * F);
    acc0 = __builtin_amdgcn_wmma_f32_16x16x32_bf16(false, a, false, b0, (short)0, acc0, false, false);
    acc1 = __builtin_amdgcn_wmma_f32_16x16x32_bf16(false, a, false, b1, (short)0, acc1, false, false);
    acc2 = __builtin_amdgcn_wmma_f32_16x16x32_bf16(false, a, false, b2, (short)0, acc2, false, false);
    acc3 = __builtin_amdgcn_wmma_f32_16x16x32_bf16(false, a, false, b3, (short)0, acc3, false, false);
  }
#endif
  // D layout: VGPR r -> row = r + hi*8, col = n0 + (lane&15)
  float* hp = H + (size_t)(row_tile * 16 + (hi << 3)) * F + ngrp * 64 + m;
  #pragma unroll
  for (int r = 0; r < 8; ++r) {
    hp[r * F +  0] = acc0[r];
    hp[r * F + 16] = acc1[r];
    hp[r * F + 32] = acc2[r];
    hp[r * F + 48] = acc3[r];
  }
}

// per (node, head): alpha_src / alpha_dst dot products over 32 channels
__global__ void alpha_kernel(const float* __restrict__ h,
                             const float* __restrict__ att_s, const float* __restrict__ att_d,
                             float* __restrict__ as_, float* __restrict__ ad_, int N) {
  int t = blockIdx.x * blockDim.x + threadIdx.x;
  if (t >= N * HEADS) return;
  int n = t >> 2, head = t & 3;
  const float* hp = h + (size_t)n * F + head * 32;
  const float* ws = att_s + head * 32;
  const float* wd = att_d + head * 32;
  float s = 0.f, d = 0.f;
  #pragma unroll
  for (int c = 0; c < 32; ++c) { float v = hp[c]; s = fmaf(v, ws[c], s); d = fmaf(v, wd[c], d); }
  as_[t] = s; ad_[t] = d;
}

// pass 1: segment-max of leaky_relu(alpha_s[src]+alpha_d[dst]) per dst,head
__global__ void edge_max_kernel(const int* __restrict__ ei, int E, int N,
                                const float* __restrict__ as_, const float* __restrict__ ad_,
                                float* __restrict__ mb) {
  int t = blockIdx.x * blockDim.x + threadIdx.x;
  int edge = t >> 2;
  if (edge >= E + N) return;
  int head = t & 3;
  int s, d;
  if (edge < E) { s = ei[edge]; d = ei[E + edge]; } else { s = edge - E; d = s; }
  float e = lrelu(as_[s * HEADS + head] + ad_[d * HEADS + head]);
  atomicMax(mb + d * HEADS + head, e);
}

// pass 2: segment-sum of exp(e - max)
__global__ void edge_sum_kernel(const int* __restrict__ ei, int E, int N,
                                const float* __restrict__ as_, const float* __restrict__ ad_,
                                const float* __restrict__ mb, float* __restrict__ sb) {
  int t = blockIdx.x * blockDim.x + threadIdx.x;
  int edge = t >> 2;
  if (edge >= E + N) return;
  int head = t & 3;
  int s, d;
  if (edge < E) { s = ei[edge]; d = ei[E + edge]; } else { s = edge - E; d = s; }
  float e = lrelu(as_[s * HEADS + head] + ad_[d * HEADS + head]);
  atomicAdd(sb + d * HEADS + head, __expf(e - mb[d * HEADS + head]));
}

// pass 3: out[dst] += h[src] * a ; one wave per edge, 4 channels per lane (float4 gather)
__global__ void __launch_bounds__(256) edge_scatter_kernel(const int* __restrict__ ei, int E, int N,
                                   const float* __restrict__ h,
                                   const float* __restrict__ as_, const float* __restrict__ ad_,
                                   const float* __restrict__ mb, const float* __restrict__ sb,
                                   float* __restrict__ out) {
  int wave = blockIdx.x * (blockDim.x >> 5) + (threadIdx.x >> 5);
  if (wave >= E + N) return;
  int lane = threadIdx.x & 31;
  int s, d;
  if (wave < E) { s = ei[wave]; d = ei[E + wave]; } else { s = wave - E; d = s; }
  int c = lane << 2;       // 4 consecutive channels
  int head = lane >> 3;    // (lane*4)/32
  float e = lrelu(as_[s * HEADS + head] + ad_[d * HEADS + head]);
  float a = __expf(e - mb[d * HEADS + head]) / (sb[d * HEADS + head] + EPS_A);
  const float4 hv = *(const float4*)(h + (size_t)s * F + c);
  float* o = out + (size_t)d * F + c;
  atomicAdd(o + 0, hv.x * a);
  atomicAdd(o + 1, hv.y * a);
  atomicAdd(o + 2, hv.z * a);
  atomicAdd(o + 3, hv.w * a);
}

__global__ void bias_kernel(float* __restrict__ buf, const float* __restrict__ b, int N) {
  int t = blockIdx.x * blockDim.x + threadIdx.x;
  if (t < N * F) buf[t] += b[t & (F - 1)];
}

// MLP head: concat(x3,x0) @ W1 + b1 -> relu -> @ W2 + b2 -> sigmoid
__global__ void mlp_kernel(const float* __restrict__ x3, const float* __restrict__ x0,
                           const float* __restrict__ W1, const float* __restrict__ b1,
                           const float* __restrict__ W2, const float* __restrict__ b2,
                           float* __restrict__ out, int N) {
  int n = blockIdx.x * blockDim.x + threadIdx.x;
  if (n >= N) return;
  float acc[32];
  #pragma unroll
  for (int j = 0; j < 32; ++j) acc[j] = b1[j];
  const float* xa = x3 + (size_t)n * F;
  for (int k = 0; k < F; ++k) {
    float xv = xa[k];
    const float* wr = W1 + k * 32;
    #pragma unroll
    for (int j = 0; j < 32; ++j) acc[j] = fmaf(xv, wr[j], acc[j]);
  }
  const float* xc = x0 + (size_t)n * F;
  for (int k = 0; k < F; ++k) {
    float xv = xc[k];
    const float* wr = W1 + (F + k) * 32;
    #pragma unroll
    for (int j = 0; j < 32; ++j) acc[j] = fmaf(xv, wr[j], acc[j]);
  }
  float o = b2[0];
  #pragma unroll
  for (int j = 0; j < 32; ++j) { float r = acc[j] > 0.f ? acc[j] : 0.f; o = fmaf(r, W2[j], o); }
  out[n] = 1.f / (1.f + __expf(-o));
}

extern "C" void kernel_launch(void* const* d_in, const int* in_sizes, int n_in,
                              void* d_out, int out_size, void* d_ws, size_t ws_size,
                              hipStream_t stream) {
  const float* x       = (const float*)d_in[0];
  const int*   ei      = (const int*)d_in[1];
  const float* W       = (const float*)d_in[2];
  const float* att_src = (const float*)d_in[3];
  const float* att_dst = (const float*)d_in[4];
  const float* b_conv  = (const float*)d_in[5];
  const float* W1      = (const float*)d_in[6];
  const float* b1      = (const float*)d_in[7];
  const float* W2      = (const float*)d_in[8];
  const float* b2      = (const float*)d_in[9];

  const int N = in_sizes[0] / F;
  const int E = in_sizes[1] / 2;
  const int Etot = E + N;

  // workspace layout (~67 MB): node buffer reused as both layer input and output
  float*  buf = (float*)d_ws;                          // N*F fp32 (x_cur / out accum)
  __bf16* xb  = (__bf16*)(buf + (size_t)N * F);        // N*F bf16
  __bf16* WbT = xb + (size_t)N * F;                    // F*F bf16 (transposed)
  float*  h   = (float*)(WbT + F * F);                 // N*F fp32
  float*  as_ = h + (size_t)N * F;                     // N*4
  float*  ad_ = as_ + (size_t)N * HEADS;               // N*4
  float*  mb  = ad_ + (size_t)N * HEADS;               // N*4 segment max
  float*  sb  = mb + (size_t)N * HEADS;                // N*4 segment sum

  const int TB = 256;
  dim3 blk(TB);
  int gNF   = (N * F + TB - 1) / TB;
  int gNH   = (N * HEADS + TB - 1) / TB;
  int gEH   = (Etot * HEADS + TB - 1) / TB;
  int gW    = (F * F + TB - 1) / TB;
  int waves_gemm = (N >> 4) * 2;
  int gGemm = (waves_gemm + (TB / 32) - 1) / (TB / 32);
  int gScat = (Etot + (TB / 32) - 1) / (TB / 32);
  int gMlp  = (N + TB - 1) / TB;

  for (int l = 0; l < 3; ++l) {
    const float* xin = (l == 0) ? x : buf;
    convert_x_kernel<<<gNF, blk, 0, stream>>>(xin, xb, N * F);
    convert_w_kernel<<<gW, blk, 0, stream>>>(W + (size_t)l * F * F, WbT);
    init_kernel<<<gNF, blk, 0, stream>>>(buf, mb, sb, N);  // after convert: buf is dead as input
    gemm_wmma_kernel<<<gGemm, blk, 0, stream>>>(xb, WbT, h, N);
    alpha_kernel<<<gNH, blk, 0, stream>>>(h, att_src + (size_t)l * F, att_dst + (size_t)l * F, as_, ad_, N);
    edge_max_kernel<<<gEH, blk, 0, stream>>>(ei, E, N, as_, ad_, mb);
    edge_sum_kernel<<<gEH, blk, 0, stream>>>(ei, E, N, as_, ad_, mb, sb);
    edge_scatter_kernel<<<gScat, blk, 0, stream>>>(ei, E, N, h, as_, ad_, mb, sb, buf);
    bias_kernel<<<gNF, blk, 0, stream>>>(buf, b_conv + (size_t)l * F, N);
  }
  mlp_kernel<<<gMlp, blk, 0, stream>>>(buf, x, W1, b1, W2, b2, (float*)d_out, N);
}
